// ToxicityLSTM_38173669327382
// MI455X (gfx1250) — matrix operand order repaired and checked by reading, not compiled, not run on hardware
//
#include <hip/hip_runtime.h>
#include <hip/hip_bf16.h>
#include <math.h>

// ---------------------------------------------------------------------------
// Bidirectional 2-layer peephole LSTM (B=32, T=128, E=H=512) for gfx1250.
// Strategy:
//   * bf16 WMMA (v_wmma_f32_16x16x32_bf16) for all GEMMs, fp32 accumulate.
//   * Input projections computed as one wide GEMM per layer (time-parallel).
//   * Recurrent part: one persistent 1024-thread block (32 wave32 waves)
//     per direction; c-state lives in registers, h ping-pongs through LDS.
// ---------------------------------------------------------------------------

typedef __bf16        v16bf __attribute__((ext_vector_type(16)));
typedef float         v8f   __attribute__((ext_vector_type(8)));
typedef unsigned int  u32x4 __attribute__((ext_vector_type(4)));
typedef unsigned short u16;

struct F2 { u32x4 lo; u32x4 hi; };   // POD 32-byte container for a bf16 frag

static __device__ inline v16bf ld_frag(const u16* p, int off0, int off1) {
  F2 f;
  f.lo = *(const u32x4*)(p + off0);
  f.hi = *(const u32x4*)(p + off1);
  return __builtin_bit_cast(v16bf, f);
}

static __device__ inline u16 f2bf(float f) {
  unsigned int u = __float_as_uint(f);
  unsigned int r = u + 0x7FFFu + ((u >> 16) & 1u);   // round-to-nearest-even
  return (u16)(r >> 16);
}
static __device__ inline float bf2f(u16 h) {
  return __uint_as_float(((unsigned int)h) << 16);
}
static __device__ inline float sigm_(float x) { return 1.f / (1.f + __expf(-x)); }
static __device__ inline float tanh_(float x) {
  x = fminf(15.f, fmaxf(-15.f, x));
  float e = __expf(2.f * x);
  return (e - 1.f) / (e + 1.f);
}

// Sizes
#define TT 128
#define BB 32
#define EE 512
#define HH 512
#define N4H 2048

// ---------------------------------------------------------------------------
// Convert + transpose weights: WT[d,l][n=0..2047][k=0..1023] = W[d,l][k][n]
// ---------------------------------------------------------------------------
__global__ void transW_kernel(const float* __restrict__ W, u16* __restrict__ WT) {
  int idx = blockIdx.x * blockDim.x + threadIdx.x;        // 4 * 2048 * 1024
  int dl  = idx >> 21;
  int rem = idx & ((1 << 21) - 1);
  int n   = rem >> 10;
  int k   = rem & 1023;
  WT[idx] = f2bf(W[(size_t)dl * 2097152 + (size_t)k * 2048 + n]);
}

// ---------------------------------------------------------------------------
// Embedding gather -> Xemb bf16 [t][b][e] (forward time order)
// ---------------------------------------------------------------------------
__global__ void embed_kernel(const int* __restrict__ tok,
                             const float* __restrict__ emb,
                             u16* __restrict__ Xemb) {
  int idx = blockIdx.x * blockDim.x + threadIdx.x;        // T*B*(E/8) = 262144
  int e8 = idx & 63;
  int b  = (idx >> 6) & 31;
  int t  = idx >> 11;
  int token = tok[b * TT + t];
  const float* src = emb + (size_t)token * EE + e8 * 8;
  unsigned int w0 = (unsigned)f2bf(src[0]) | ((unsigned)f2bf(src[1]) << 16);
  unsigned int w1 = (unsigned)f2bf(src[2]) | ((unsigned)f2bf(src[3]) << 16);
  unsigned int w2 = (unsigned)f2bf(src[4]) | ((unsigned)f2bf(src[5]) << 16);
  unsigned int w3 = (unsigned)f2bf(src[6]) | ((unsigned)f2bf(src[7]) << 16);
  u32x4 v = {w0, w1, w2, w3};
  *(u32x4*)(Xemb + (size_t)(t * BB + b) * EE + e8 * 8) = v;
}

// ---------------------------------------------------------------------------
// Stage A: xproj[dir][m=t*32+b][4H] = In[m] @ Wx  (fp32 out, bf16 WMMA)
// grid = (32 mblocks, 16 nblocks, 2 dirs), 256 threads (8 waves).
// Wave w computes a 16x128 strip: 8 WMMA tiles, K = 512 in 16 chunks of 32.
// ---------------------------------------------------------------------------
__global__ __launch_bounds__(256) void gemm_xproj_kernel(
    const u16* __restrict__ Xemb, const u16* __restrict__ h_all,
    const u16* __restrict__ WT, float* __restrict__ xproj, int layer) {
  const int dir    = blockIdx.z;
  const int nblock = blockIdx.y;
  const int mblock = blockIdx.x;
  const int tid    = threadIdx.x;
  const int wave   = tid >> 5;
  const int lane   = tid & 31;
  const int laneN  = lane & 15;
  const int laneHi = lane >> 4;
  const int mtile  = mblock * 8 + wave;      // 0..255
  const int mrow   = mtile * 16 + laneN;     // A-row read by this lane
  const int t = mrow >> 5, b = mrow & 31;

  const u16* In;
  if (layer == 0) {
    int srow = (dir == 1) ? ((TT - 1 - t) * BB + b) : mrow;  // bw: reversed time
    In = Xemb + (size_t)srow * EE;
  } else {
    In = h_all + (size_t)dir * TT * BB * HH + (size_t)mrow * HH;
  }
  const u16* WTb = WT + (size_t)(dir * 2 + layer) * N4H * 1024;  // x-part: k 0..511
  float* xpd = xproj + (size_t)dir * TT * BB * N4H;

  v8f acc[8];
  for (int nt = 0; nt < 8; ++nt)
    for (int j = 0; j < 8; ++j) acc[nt][j] = 0.f;

  for (int kc = 0; kc < 16; ++kc) {
    int ka = kc * 32 + laneHi * 8;
    v16bf a = ld_frag(In, ka, ka + 16);
    int kb = kc * 32 + laneHi * 8;
    for (int nt = 0; nt < 8; ++nt) {
      int n = nblock * 128 + nt * 16 + laneN;
      v16bf bm = ld_frag(WTb + (size_t)n * 1024, kb, kb + 16);
      acc[nt] = __builtin_amdgcn_wmma_f32_16x16x32_bf16(
          false, a, false, bm, (short)0, acc[nt], false, false);
    }
  }
  for (int nt = 0; nt < 8; ++nt) {
    int n = nblock * 128 + nt * 16 + laneN;
    for (int j = 0; j < 8; ++j) {
      int m = mtile * 16 + j + laneHi * 8;
      xpd[(size_t)m * N4H + n] = acc[nt][j];
    }
  }
}

// ---------------------------------------------------------------------------
// Recurrent scan: one block per direction, 1024 threads = 32 waves.
// Wave w owns h-column tile ht=w; it computes i,j,f,o tiles for batch tiles
// m=0 (rows 0..15) and m=1 (rows 16..31), so the gate math is in-register.
// Cell state c stays in VGPRs across all 128 steps; h goes through LDS.
// ---------------------------------------------------------------------------
__global__ __launch_bounds__(1024) void lstm_scan_kernel(
    const u16* __restrict__ WT, const float* __restrict__ xproj,
    u16* __restrict__ h_all, const float* __restrict__ bias,
    const float* __restrict__ peep, int layer) {
  __shared__ __align__(16) u16 hs[BB * HH];     // 32 KB

  const int dir    = blockIdx.x;
  const int tid    = threadIdx.x;
  const int wave   = tid >> 5;
  const int lane   = tid & 31;
  const int laneN  = lane & 15;
  const int laneHi = lane >> 4;
  const int ht     = wave;                      // 0..31 h-column tile
  const int hcol   = ht * 16 + laneN;

  const u16*  WhT  = WT + (size_t)(dir * 2 + layer) * N4H * 1024 + 512;  // h-part
  const float* xp  = xproj + (size_t)dir * TT * BB * N4H;
  u16*   hout      = h_all + (size_t)dir * TT * BB * HH;
  const float* bv  = bias + (dir * 2 + layer) * N4H;
  const float* pv  = peep + (dir * 2 + layer) * 3 * HH;

  const float pi = pv[hcol], pf = pv[HH + hcol], po = pv[2 * HH + hcol];
  const float bi = bv[hcol], bj = bv[HH + hcol];
  const float bff = bv[2 * HH + hcol], bo = bv[3 * HH + hcol];

  for (int i = tid; i < BB * HH; i += 1024) hs[i] = 0;
  v8f cst[2];
  for (int m = 0; m < 2; ++m)
    for (int j = 0; j < 8; ++j) cst[m][j] = 0.f;
  __syncthreads();

  for (int t = 0; t < TT; ++t) {
    // z init = input projection (bias added later)
    v8f acc[2][4];
    for (int m = 0; m < 2; ++m)
      for (int g = 0; g < 4; ++g)
        for (int j = 0; j < 8; ++j)
          acc[m][g][j] =
              xp[(size_t)(t * BB + m * 16 + j + laneHi * 8) * N4H + g * HH + hcol];

    // z += h_prev @ Wh   (K = 512, 16 chunks of 32)
    for (int kc = 0; kc < 16; ++kc) {
      int ka = kc * 32 + laneHi * 8;
      v16bf a0 = ld_frag(hs + (size_t)(laneN) * HH, ka, ka + 16);
      v16bf a1 = ld_frag(hs + (size_t)(16 + laneN) * HH, ka, ka + 16);
      int kb = kc * 32 + laneHi * 8;
      for (int g = 0; g < 4; ++g) {
        v16bf bm = ld_frag(WhT + (size_t)(g * HH + hcol) * 1024, kb, kb + 16);
        acc[0][g] = __builtin_amdgcn_wmma_f32_16x16x32_bf16(
            false, a0, false, bm, (short)0, acc[0][g], false, false);
        acc[1][g] = __builtin_amdgcn_wmma_f32_16x16x32_bf16(
            false, a1, false, bm, (short)0, acc[1][g], false, false);
      }
    }
    __syncthreads();   // all waves done reading h_prev from LDS

    for (int m = 0; m < 2; ++m) {
      for (int j = 0; j < 8; ++j) {
        float zi = acc[m][0][j] + bi;
        float zj = acc[m][1][j] + bj;
        float zf = acc[m][2][j] + bff + 1.0f;   // FORGET_BIAS
        float zo = acc[m][3][j] + bo;
        float cc = cst[m][j];
        float iv = sigm_(zi + pi * cc);
        float fv = sigm_(zf + pf * cc);
        float cn = fv * cc + iv * tanh_(zj);
        float ov = sigm_(zo + po * cn);
        float hn = ov * tanh_(cn);
        cst[m][j] = cn;
        int brow = m * 16 + j + laneHi * 8;
        u16 hb = f2bf(hn);
        hs[brow * HH + hcol] = hb;
        hout[(size_t)(t * BB + brow) * HH + hcol] = hb;
      }
    }
    __syncthreads();   // new h visible to all waves
  }
}

// ---------------------------------------------------------------------------
// Final: reproduce the reshape bug. last[b] = bw layer-1 h at stack index
// 64 (even b, batch b/2) or 0 (odd b, batch (b-1)/2). Then logits/sigmoid.
// ---------------------------------------------------------------------------
__global__ void final_kernel(const u16* __restrict__ h_all,
                             const float* __restrict__ w_out,
                             const float* __restrict__ b_out,
                             float* __restrict__ out) {
  int tid = threadIdx.x;
  if (tid >= BB * 6) return;
  int b = tid / 6, l = tid % 6;
  int m    = b >> 1;
  int sidx = (b & 1) ? 0 : 64;
  const u16* hp = h_all + (size_t)1 * TT * BB * HH + (size_t)(sidx * BB + m) * HH;
  float s = b_out[l];
  for (int k = 0; k < HH; ++k) s += bf2f(hp[k]) * w_out[k * 6 + l];
  float o = 1.0f / (1.0f + __expf(-s));
  float p = (o > 0.5f) ? 1.0f : 0.0f;
  out[b * 6 + l]            = s;   // logits
  out[BB * 6 + b * 6 + l]   = o;   // output
  out[2 * BB * 6 + b * 6 + l] = p; // prediction
}

// ---------------------------------------------------------------------------
extern "C" void kernel_launch(void* const* d_in, const int* in_sizes, int n_in,
                              void* d_out, int out_size, void* d_ws, size_t ws_size,
                              hipStream_t stream) {
  (void)in_sizes; (void)n_in; (void)out_size; (void)ws_size;
  const int*   tok   = (const int*)d_in[0];
  const float* emb   = (const float*)d_in[1];
  const float* W     = (const float*)d_in[2];
  const float* bias  = (const float*)d_in[3];
  const float* peep  = (const float*)d_in[4];
  const float* w_out = (const float*)d_in[5];
  const float* b_out = (const float*)d_in[6];
  float* out = (float*)d_out;

  // Workspace layout (16B-aligned):
  //   WT    bf16 [2][2][2048][1024]            16 MB
  //   Xemb  bf16 [T][B][E]                      4 MB
  //   xproj f32  [2][T*B][4H]                  64 MB
  //   h_all bf16 [2][T][B][H]                   8 MB
  u16*   WT    = (u16*)d_ws;
  u16*   Xemb  = WT + (size_t)8388608;
  float* xproj = (float*)(Xemb + (size_t)2097152);
  u16*   h_all = (u16*)(xproj + (size_t)16777216);

  transW_kernel<<<32768, 256, 0, stream>>>(W, WT);
  embed_kernel<<<1024, 256, 0, stream>>>(tok, emb, Xemb);

  gemm_xproj_kernel<<<dim3(32, 16, 2), 256, 0, stream>>>(Xemb, h_all, WT, xproj, 0);
  lstm_scan_kernel<<<2, 1024, 0, stream>>>(WT, xproj, h_all, bias, peep, 0);
  gemm_xproj_kernel<<<dim3(32, 16, 2), 256, 0, stream>>>(Xemb, h_all, WT, xproj, 1);
  lstm_scan_kernel<<<2, 1024, 0, stream>>>(WT, xproj, h_all, bias, peep, 1);

  final_kernel<<<1, 192, 0, stream>>>(h_all, w_out, b_out, out);
}
